// DGCNN_6339371729770
// MI455X (gfx1250) — matrix operand order repaired and checked by reading, not compiled, not run on hardware
//
#include <hip/hip_runtime.h>
#include <cstdint>
#include <cstddef>

// ---------------------------------------------------------------------------
// CDNA5 / gfx1250 DGCNN implementation.
// Precision strategy: f32 inputs split as x = hi + lo (bf16 each);
// dot(x,y) ~= hi*hi + hi*lo + lo*hi accumulated in f32 via
// v_wmma_f32_16x16x32_bf16  (error ~2^-16 relative, ~f32-equivalent for knn).
// ---------------------------------------------------------------------------

typedef __bf16 bf16_t;
typedef __attribute__((ext_vector_type(16))) __bf16 v16bf;
typedef __attribute__((ext_vector_type(8)))  __bf16 v8bf;
typedef __attribute__((ext_vector_type(8)))  float  v8f;

#define KNN_K 7
#define KNN_WAVES 4

__device__ __forceinline__ v8f wmma_bf16(v16bf a, v16bf b, v8f c) {
    // 8-arg form: (neg_a, A, neg_b, B, c_mod, C, reuse_a, reuse_b)
    return __builtin_amdgcn_wmma_f32_16x16x32_bf16(false, a, false, b,
                                                   (short)0, c, false, false);
}

// A-matrix fragment, 16x32 bf16 (M x K), rows are row-major in memory.
// ISA 7.12.2: lanes 0-15 -> M=lane, K = {kc+0..7, kc+16..23}
//             lanes16-31 -> M=lane-16, K = {kc+8..15, kc+24..31}
__device__ __forceinline__ v16bf load_a_frag(const bf16_t* __restrict__ base,
                                             int rowbase, int ld, int kc, int lane) {
    int r    = rowbase + (lane & 15);
    int koff = kc + ((lane >> 4) << 3);          // +8 for upper half-wave
    const bf16_t* p = base + (size_t)r * ld + koff;
    v8bf lo8 = *(const v8bf*)(p);                // K = koff .. koff+7
    v8bf hi8 = *(const v8bf*)(p + 16);           // K = koff+16 .. koff+23
    v16bf out;
#pragma unroll
    for (int i = 0; i < 8; ++i) { out[i] = lo8[i]; out[8 + i] = hi8[i]; }
    return out;
}

// B-matrix fragment, 32x16 bf16 (K x N); column n of B = row (colbase+n) of a
// row-major matrix, contiguous along K.
// ISA 7.12.2: lanes 0-15 -> N=lane,   K = {kc+0..7, kc+16..23}
//             lanes16-31 -> N=lane-16, K = {kc+8..15, kc+24..31}
__device__ __forceinline__ v16bf load_b_frag(const bf16_t* __restrict__ base,
                                             int colbase, int ld, int kc, int lane) {
    int c    = colbase + (lane & 15);
    int koff = kc + ((lane >> 4) << 3);          // +8 for upper half-wave
    const bf16_t* p = base + (size_t)c * ld + koff;
    v8bf lo8 = *(const v8bf*)(p);
    v8bf hi8 = *(const v8bf*)(p + 16);
    v16bf out;
#pragma unroll
    for (int i = 0; i < 8; ++i) { out[i] = lo8[i]; out[8 + i] = hi8[i]; }
    return out;
}

// ---------------------------------------------------------------------------
// Split f32 -> (hi, lo) bf16 pair.
// ---------------------------------------------------------------------------
__global__ void dgcnn_split_kernel(const float* __restrict__ in,
                                   bf16_t* __restrict__ hi, bf16_t* __restrict__ lo,
                                   long n) {
    long i = (long)blockIdx.x * blockDim.x + threadIdx.x;
    if (i < n) {
        float f = in[i];
        bf16_t h = (bf16_t)f;
        hi[i] = h;
        lo[i] = (bf16_t)(f - (float)h);
    }
}

// Row squared norms (one block per row, blockDim == C <= 256).
__global__ void dgcnn_sqnorm_kernel(const float* __restrict__ x,
                                    float* __restrict__ sq, int C) {
    __shared__ float red[256];
    int n = blockIdx.x, c = threadIdx.x;
    float v = x[(size_t)n * C + c];
    red[c] = v * v;
    __syncthreads();
    for (int s = blockDim.x >> 1; s > 0; s >>= 1) {
        if (c < s) red[c] += red[c + s];
        __syncthreads();
    }
    if (c == 0) sq[n] = red[0];
}

// ---------------------------------------------------------------------------
// Shared top-7 selection epilogue for a 32x16 score tile staged in LDS.
// ---------------------------------------------------------------------------
__device__ __forceinline__ void knn_select_tile(float (*dtw)[17], const float* __restrict__ sq,
                                                int jb, int lane, float* bd, int* bi) {
    // In-wave cross-lane LDS RAW: DS ops are in-order per wave; force
    // completion + block compiler reordering.
    asm volatile("s_wait_dscnt 0" ::: "memory");
#pragma unroll
    for (int t = 0; t < 16; ++t) {
        int   j = jb + t;
        float s = sq[j] - 2.0f * dtw[lane][t];
        if (s < bd[KNN_K - 1]) {
            bd[KNN_K - 1] = s; bi[KNN_K - 1] = j;
#pragma unroll
            for (int q = KNN_K - 1; q > 0; --q) {
                if (bd[q] < bd[q - 1]) {
                    float td = bd[q]; bd[q] = bd[q - 1]; bd[q - 1] = td;
                    int   ti = bi[q]; bi[q] = bi[q - 1]; bi[q - 1] = ti;
                }
            }
        }
    }
    asm volatile("s_wait_dscnt 0" ::: "memory");  // reads done before tile rewrite
}

// ---------------------------------------------------------------------------
// Fused kNN, compile-time C: each wave owns 32 rows (two 16x16 WMMA C-tiles).
// A-fragments for the wave's rows are hoisted into registers ONCE, then the
// wave sweeps all N columns in 16-wide tiles (B-fragment loads only in the
// steady state: 6 WMMA per 4 b128 loads per K-chunk).
// Scores = sq[j] - 2*dot (row-constant sq[i] dropped: cannot change ordering).
// ---------------------------------------------------------------------------
template <int C>
__global__ __launch_bounds__(32 * KNN_WAVES)
void dgcnn_knn_reg_kernel(const bf16_t* __restrict__ Xhi, const bf16_t* __restrict__ Xlo,
                          const float* __restrict__ sq, int N,
                          int* __restrict__ idx_out) {
    constexpr int KC = C / 32;
    __shared__ float dt[KNN_WAVES][32][17];      // padded: no bank conflicts
    int lane = threadIdx.x & 31;
    int wave = threadIdx.x >> 5;
    int rowbase = (blockIdx.x * KNN_WAVES + wave) * 32;
    if (rowbase >= N) return;

    float bd[KNN_K];
    int   bi[KNN_K];
#pragma unroll
    for (int t = 0; t < KNN_K; ++t) { bd[t] = 3.4e38f; bi[t] = 0; }

    int n16  = lane & 15;
    int moff = (lane >> 4) * 8;

    // Hoist loop-invariant A fragments (this wave's 32 rows) into registers.
    v16bf a0h[KC], a0l[KC], a1h[KC], a1l[KC];
#pragma unroll
    for (int k = 0; k < KC; ++k) {
        a0h[k] = load_a_frag(Xhi, rowbase,      C, k * 32, lane);
        a0l[k] = load_a_frag(Xlo, rowbase,      C, k * 32, lane);
        a1h[k] = load_a_frag(Xhi, rowbase + 16, C, k * 32, lane);
        a1l[k] = load_a_frag(Xlo, rowbase + 16, C, k * 32, lane);
    }

    for (int jb = 0; jb < N; jb += 16) {
        v8f c0 = {}, c1 = {};
#pragma unroll
        for (int k = 0; k < KC; ++k) {
            v16bf bh = load_b_frag(Xhi, jb, C, k * 32, lane);
            v16bf bl = load_b_frag(Xlo, jb, C, k * 32, lane);
            c0 = wmma_bf16(a0h[k], bh, c0);
            c0 = wmma_bf16(a0h[k], bl, c0);
            c0 = wmma_bf16(a0l[k], bh, c0);
            c1 = wmma_bf16(a1h[k], bh, c1);
            c1 = wmma_bf16(a1h[k], bl, c1);
            c1 = wmma_bf16(a1l[k], bh, c1);
        }
#pragma unroll
        for (int r = 0; r < 8; ++r) {
            dt[wave][moff + r][n16]      = c0[r];
            dt[wave][16 + moff + r][n16] = c1[r];
        }
        knn_select_tile(dt[wave], sq, jb, lane, bd, bi);
    }

    int row = rowbase + lane;
#pragma unroll
    for (int t = 0; t < KNN_K; ++t) idx_out[(size_t)row * KNN_K + t] = bi[t];
}

// Dynamic-C fallback (same algorithm, A reloaded per tile).
__global__ __launch_bounds__(32 * KNN_WAVES)
void dgcnn_knn_dyn_kernel(const bf16_t* __restrict__ Xhi, const bf16_t* __restrict__ Xlo,
                          const float* __restrict__ sq, int N, int C,
                          int* __restrict__ idx_out) {
    __shared__ float dt[KNN_WAVES][32][17];
    int lane = threadIdx.x & 31;
    int wave = threadIdx.x >> 5;
    int rowbase = (blockIdx.x * KNN_WAVES + wave) * 32;
    if (rowbase >= N) return;

    float bd[KNN_K];
    int   bi[KNN_K];
#pragma unroll
    for (int t = 0; t < KNN_K; ++t) { bd[t] = 3.4e38f; bi[t] = 0; }

    int n16  = lane & 15;
    int moff = (lane >> 4) * 8;

    for (int jb = 0; jb < N; jb += 16) {
        v8f c0 = {}, c1 = {};
        for (int kc = 0; kc < C; kc += 32) {
            v16bf bh  = load_b_frag(Xhi, jb, C, kc, lane);
            v16bf bl  = load_b_frag(Xlo, jb, C, kc, lane);
            v16bf a0h = load_a_frag(Xhi, rowbase,      C, kc, lane);
            v16bf a0l = load_a_frag(Xlo, rowbase,      C, kc, lane);
            v16bf a1h = load_a_frag(Xhi, rowbase + 16, C, kc, lane);
            v16bf a1l = load_a_frag(Xlo, rowbase + 16, C, kc, lane);
            c0 = wmma_bf16(a0h, bh, c0);
            c0 = wmma_bf16(a0h, bl, c0);
            c0 = wmma_bf16(a0l, bh, c0);
            c1 = wmma_bf16(a1h, bh, c1);
            c1 = wmma_bf16(a1h, bl, c1);
            c1 = wmma_bf16(a1l, bh, c1);
        }
#pragma unroll
        for (int r = 0; r < 8; ++r) {
            dt[wave][moff + r][n16]      = c0[r];
            dt[wave][16 + moff + r][n16] = c1[r];
        }
        knn_select_tile(dt[wave], sq, jb, lane, bd, bi);
    }

    int row = rowbase + lane;
#pragma unroll
    for (int t = 0; t < KNN_K; ++t) idx_out[(size_t)row * KNN_K + t] = bi[t];
}

// ---------------------------------------------------------------------------
// Y[N,H] = X[N,C] @ W[:, coff:coff+C]^T   (W row-major, row stride ldw)
// One wave per block: 32x16 output tile.
// ---------------------------------------------------------------------------
__global__ __launch_bounds__(32)
void dgcnn_gemm_kernel(const bf16_t* __restrict__ Xhi, const bf16_t* __restrict__ Xlo,
                       const bf16_t* __restrict__ Whi, const bf16_t* __restrict__ Wlo,
                       float* __restrict__ Y, int N, int C, int H, int ldw, int coff) {
    int lane    = threadIdx.x;
    int rowbase = blockIdx.x * 32;
    int hbase   = blockIdx.y * 16;
    v8f c0 = {}, c1 = {};
    for (int kc = 0; kc < C; kc += 32) {
        v16bf bh  = load_b_frag(Whi + coff, hbase, ldw, kc, lane);
        v16bf bl  = load_b_frag(Wlo + coff, hbase, ldw, kc, lane);
        v16bf a0h = load_a_frag(Xhi, rowbase,      C, kc, lane);
        v16bf a0l = load_a_frag(Xlo, rowbase,      C, kc, lane);
        v16bf a1h = load_a_frag(Xhi, rowbase + 16, C, kc, lane);
        v16bf a1l = load_a_frag(Xlo, rowbase + 16, C, kc, lane);
        c0 = wmma_bf16(a0h, bh, c0);
        c0 = wmma_bf16(a0h, bl, c0);
        c0 = wmma_bf16(a0l, bh, c0);
        c1 = wmma_bf16(a1h, bh, c1);
        c1 = wmma_bf16(a1h, bl, c1);
        c1 = wmma_bf16(a1l, bh, c1);
    }
    int n16  = lane & 15;
    int moff = (lane >> 4) * 8;
#pragma unroll
    for (int r = 0; r < 8; ++r) {
        Y[(size_t)(rowbase + moff + r) * H + hbase + n16]      = c0[r];
        Y[(size_t)(rowbase + 16 + moff + r) * H + hbase + n16] = c1[r];
    }
}

// ---------------------------------------------------------------------------
// Edge-conv epilogue: h[n,c] = u[n,c] - v[n,c] + b[c] + max_k v[idx[n,k], c]
// Writes bf16 hi/lo split of h plus its row sq-norm (for next knn stage).
// One block per row, blockDim == H.
// ---------------------------------------------------------------------------
__global__ void dgcnn_epilogue1_kernel(const float* __restrict__ u, const float* __restrict__ v,
                                       const float* __restrict__ b, const int* __restrict__ idx,
                                       int H, bf16_t* __restrict__ Hhi, bf16_t* __restrict__ Hlo,
                                       float* __restrict__ sqh) {
    __shared__ float red[256];
    int n = blockIdx.x, c = threadIdx.x;
    float mv = -3.4e38f;
#pragma unroll
    for (int t = 0; t < KNN_K; ++t) {
        int j = idx[(size_t)n * KNN_K + t];
        mv = fmaxf(mv, v[(size_t)j * H + c]);
    }
    float h = u[(size_t)n * H + c] - v[(size_t)n * H + c] + b[c] + mv;
    bf16_t hh = (bf16_t)h;
    Hhi[(size_t)n * H + c] = hh;
    Hlo[(size_t)n * H + c] = (bf16_t)(h - (float)hh);
    red[c] = h * h;
    __syncthreads();
    for (int s = H >> 1; s > 0; s >>= 1) {
        if (c < s) red[c] += red[c + s];
        __syncthreads();
    }
    if (c == 0) sqh[n] = red[0];
}

// Final epilogue: out[n] = bd + sum_c Wd[c] * (u-v+b[c]+max_k v[j,c])
__global__ void dgcnn_epilogue2_kernel(const float* __restrict__ u, const float* __restrict__ v,
                                       const float* __restrict__ b, const int* __restrict__ idx,
                                       const float* __restrict__ Wd, const float* __restrict__ bd,
                                       int H, float* __restrict__ out) {
    __shared__ float red[256];
    int n = blockIdx.x, c = threadIdx.x;
    float mv = -3.4e38f;
#pragma unroll
    for (int t = 0; t < KNN_K; ++t) {
        int j = idx[(size_t)n * KNN_K + t];
        mv = fmaxf(mv, v[(size_t)j * H + c]);
    }
    float h = u[(size_t)n * H + c] - v[(size_t)n * H + c] + b[c] + mv;
    red[c] = h * Wd[c];
    __syncthreads();
    for (int s = H >> 1; s > 0; s >>= 1) {
        if (c < s) red[c] += red[c + s];
        __syncthreads();
    }
    if (c == 0) out[n] = red[0] + bd[0];
}

// ---------------------------------------------------------------------------
extern "C" void kernel_launch(void* const* d_in, const int* in_sizes, int n_in,
                              void* d_out, int out_size, void* d_ws, size_t ws_size,
                              hipStream_t stream) {
    const float* x  = (const float*)d_in[0];
    const float* W1 = (const float*)d_in[1];
    const float* b1 = (const float*)d_in[2];
    const float* W2 = (const float*)d_in[3];
    const float* b2 = (const float*)d_in[4];
    const float* Wd = (const float*)d_in[5];
    const float* bd = (const float*)d_in[6];

    const int H    = in_sizes[2];            // 256
    const int twoD = in_sizes[1] / H;        // 256
    const int d    = twoD / 2;               // 128
    const int N    = in_sizes[0] / d;        // 8192

    // Workspace carve-out (256B aligned).
    char* p = (char*)d_ws;
    auto carve = [&](size_t bytes) -> void* {
        void* r = (void*)p;
        p += (bytes + 255) & ~(size_t)255;
        return r;
    };
    bf16_t* xhi  = (bf16_t*)carve((size_t)N * d * 2);
    bf16_t* xlo  = (bf16_t*)carve((size_t)N * d * 2);
    bf16_t* w1hi = (bf16_t*)carve((size_t)H * twoD * 2);
    bf16_t* w1lo = (bf16_t*)carve((size_t)H * twoD * 2);
    bf16_t* w2hi = (bf16_t*)carve((size_t)H * 2 * H * 2);
    bf16_t* w2lo = (bf16_t*)carve((size_t)H * 2 * H * 2);
    float*  sq   = (float*)carve((size_t)N * 4);
    int*    idx1 = (int*)carve((size_t)N * KNN_K * 4);
    float*  u    = (float*)carve((size_t)N * H * 4);
    float*  v    = (float*)carve((size_t)N * H * 4);
    bf16_t* hhi  = (bf16_t*)carve((size_t)N * H * 2);
    bf16_t* hlo  = (bf16_t*)carve((size_t)N * H * 2);
    float*  sqh  = (float*)carve((size_t)N * 4);
    int*    idx2 = (int*)carve((size_t)N * KNN_K * 4);

    // --- precision splits ---
    long nx = (long)N * d, nw1 = (long)H * twoD, nw2 = (long)H * 2 * H;
    dgcnn_split_kernel<<<(nx  + 255) / 256, 256, 0, stream>>>(x,  xhi,  xlo,  nx);
    dgcnn_split_kernel<<<(nw1 + 255) / 256, 256, 0, stream>>>(W1, w1hi, w1lo, nw1);
    dgcnn_split_kernel<<<(nw2 + 255) / 256, 256, 0, stream>>>(W2, w2hi, w2lo, nw2);
    dgcnn_sqnorm_kernel<<<N, d, 0, stream>>>(x, sq, d);

    auto launch_knn = [&](const bf16_t* hi, const bf16_t* lo, const float* s,
                          int C, int* idx) {
        int blocks = N / (32 * KNN_WAVES);
        if (C == 128)
            dgcnn_knn_reg_kernel<128><<<blocks, 32 * KNN_WAVES, 0, stream>>>(hi, lo, s, N, idx);
        else if (C == 256)
            dgcnn_knn_reg_kernel<256><<<blocks, 32 * KNN_WAVES, 0, stream>>>(hi, lo, s, N, idx);
        else
            dgcnn_knn_dyn_kernel<<<blocks, 32 * KNN_WAVES, 0, stream>>>(hi, lo, s, N, C, idx);
    };

    // --- layer 1 ---
    launch_knn(xhi, xlo, sq, d, idx1);
    dgcnn_gemm_kernel<<<dim3(N / 32, H / 16), 32, 0, stream>>>(
        xhi, xlo, w1hi, w1lo, u, N, d, H, twoD, 0);
    dgcnn_gemm_kernel<<<dim3(N / 32, H / 16), 32, 0, stream>>>(
        xhi, xlo, w1hi, w1lo, v, N, d, H, twoD, d);
    dgcnn_epilogue1_kernel<<<N, H, 0, stream>>>(u, v, b1, idx1, H, hhi, hlo, sqh);

    // --- layer 2 ---
    launch_knn(hhi, hlo, sqh, H, idx2);
    dgcnn_gemm_kernel<<<dim3(N / 32, H / 16), 32, 0, stream>>>(
        hhi, hlo, w2hi, w2lo, u, N, H, H, 2 * H, 0);
    dgcnn_gemm_kernel<<<dim3(N / 32, H / 16), 32, 0, stream>>>(
        hhi, hlo, w2hi, w2lo, v, N, H, H, 2 * H, H);

    // --- head ---
    dgcnn_epilogue2_kernel<<<N, H, 0, stream>>>(u, v, b2, idx2, Wd, bd, H, (float*)d_out);
}
